// OnlineTransformer_72284299592212
// MI455X (gfx1250) — matrix operand order repaired and checked by reading, not compiled, not run on hardware
//
#include <hip/hip_runtime.h>
#include <hip/hip_bf16.h>
#include <math.h>

// ---------------------------------------------------------------------------
// Types matching the gfx1250 WMMA / TDM builtin signatures
// ---------------------------------------------------------------------------
typedef __attribute__((ext_vector_type(16))) __bf16 v16bf;
typedef __attribute__((ext_vector_type(8)))  __bf16 v8bf;
typedef __attribute__((ext_vector_type(8)))  float  v8f;
typedef __attribute__((ext_vector_type(4)))  unsigned int v4u;
typedef __attribute__((ext_vector_type(4)))  int v4i;
typedef __attribute__((ext_vector_type(8)))  int v8i;

union V16 { v16bf v; v8bf h[2]; };

#define E_DIM   480
#define F_DIM   1920
#define S_DIM   256
#define B_DIM   64
#define H_DIM   6
#define D_DIM   80
#define L_DIM   8
#define NCHORD  4576
#define MTOK    (B_DIM * S_DIM)        // 16384 rows of the residual stream
#define MH      (B_DIM * (S_DIM / 2))  // 8192 rows fed to the chord head

// Tensor Data Mover available? (device pass only; host pass falls back)
#if defined(__gfx1250__) && __has_builtin(__builtin_amdgcn_tensor_load_to_lds)
#define HAVE_TDM 1
#else
#define HAVE_TDM 0
#endif

#if HAVE_TDM
typedef __attribute__((address_space(3))) void lds_void;
__device__ __forceinline__ unsigned lds_offset(void* p) {
  // generic -> LDS addrspace cast; ptrtoint of an AS3 pointer is the LDS byte offset
  return (unsigned)(unsigned long long)(lds_void*)p;
}
#endif

__device__ __forceinline__ v8bf bzero8() {
  v8bf z;
#pragma unroll
  for (int i = 0; i < 8; ++i) z[i] = (__bf16)0.0f;
  return z;
}

// ---------------------------------------------------------------------------
// Elementwise helpers
// ---------------------------------------------------------------------------
__global__ __launch_bounds__(256) void cvt_kernel(const float* __restrict__ s,
                                                  __bf16* __restrict__ d, int n) {
  int idx = blockIdx.x * 256 + threadIdx.x;
  if (idx < n) d[idx] = (__bf16)s[idx];
}

__global__ __launch_bounds__(256) void embed_kernel(const int* __restrict__ tok,
                                                    const float* __restrict__ te,
                                                    const float* __restrict__ pe,
                                                    float* __restrict__ x) {
  size_t idx = (size_t)blockIdx.x * 256 + threadIdx.x;
  if (idx >= (size_t)MTOK * E_DIM) return;
  int m = (int)(idx / E_DIM);
  int e = (int)(idx - (size_t)m * E_DIM);
  int s = m & (S_DIM - 1);
  x[idx] = te[(size_t)tok[m] * E_DIM + e] + pe[(size_t)s * E_DIM + e];
}

__global__ __launch_bounds__(256) void gather_odd_kernel(const float* __restrict__ x,
                                                         __bf16* __restrict__ xh) {
  int idx = blockIdx.x * 256 + threadIdx.x;
  if (idx >= MH * E_DIM) return;
  int m = idx / E_DIM, e = idx - m * E_DIM;
  int b = m >> 7, t = m & 127;                 // 128 chord positions per batch
  xh[idx] = (__bf16)x[((size_t)(b * S_DIM + 2 * t + 1)) * E_DIM + e];
}

// One wave per row: 32 lanes x 15 elements, shfl_xor reductions (wave32).
__global__ __launch_bounds__(256) void ln_kernel(const float* __restrict__ x,
                                                 const float* __restrict__ g,
                                                 const float* __restrict__ bb,
                                                 __bf16* __restrict__ out) {
  const int wave = threadIdx.x >> 5, lane = threadIdx.x & 31;
  const int row = blockIdx.x * 8 + wave;
  const float* xr = x + (size_t)row * E_DIM;
  float v[15], s = 0.0f, s2 = 0.0f;
#pragma unroll
  for (int i = 0; i < 15; ++i) {
    v[i] = xr[lane + 32 * i];
    s += v[i];
    s2 += v[i] * v[i];
  }
#pragma unroll
  for (int m = 1; m < 32; m <<= 1) {
    s  += __shfl_xor(s, m);
    s2 += __shfl_xor(s2, m);
  }
  float mu   = s * (1.0f / E_DIM);
  float var  = s2 * (1.0f / E_DIM) - mu * mu;
  float rstd = rsqrtf(var + 1e-5f);
#pragma unroll
  for (int i = 0; i < 15; ++i) {
    int e = lane + 32 * i;
    out[(size_t)row * E_DIM + e] = (__bf16)((v[i] - mu) * rstd * g[e] + bb[e]);
  }
}

// ---------------------------------------------------------------------------
// WMMA GEMM: C(MxN) = A(MxK,bf16) * W(NxK,bf16)^T + bias, 5 epilogues.
// Macro-tile 128x64, 8 waves of 32x32 (2x2 WMMA tiles), BK=32.
// Double-buffered LDS (ping-pong, one barrier per k-step). The W tile is
// staged by the Tensor Data Mover when available: one tensor_load_to_lds per
// workgroup issued from wave 0 (TDM ignores EXEC), with D#.pad_interval=16DW /
// pad_amount=4DW reproducing the padded 80B LDS row layout in hardware, and
// TDM OOB-zero-fill handling the ragged N edge. A tile via b128 loads.
// ---------------------------------------------------------------------------
#define BM 128
#define BN 64
#define BK 32
#define LDT 40

template <int EPI>
__global__ __launch_bounds__(256)
void gemm_bf16(const __bf16* __restrict__ A, const __bf16* __restrict__ W,
               const float* __restrict__ bias, __bf16* __restrict__ outb,
               float* __restrict__ outf, int M, int N, int K) {
  __shared__ __attribute__((aligned(16))) __bf16 sA[2][BM * LDT];
  __shared__ __attribute__((aligned(16))) __bf16 sB[2][BN * LDT];

  const int tid  = threadIdx.x;
  const int lane = tid & 31;
  const int wave = tid >> 5;
  const int wm   = wave >> 1;   // 0..3 along M
  const int wn   = wave & 1;    // 0..1 along N
  const int half = lane >> 4;
  const int lrow = lane & 15;

  const int m0 = blockIdx.y * BM;
  const int n0 = blockIdx.x * BN;

  // A-tile staging assignment: thread -> rows {r, r+64}, 8-half chunk kc
  const int ar  = tid >> 2;
  const int akc = (tid & 3) << 3;

#if HAVE_TDM
  auto tdm_issue_B = [&](int kk, __bf16* dst) {
    // D# group 0: count=1 | lds_addr | global_addr(57b) | type=2
    unsigned long long ga =
        (unsigned long long)(const void*)(W + (size_t)n0 * K + kk);
    v4u g0;
    g0[0] = 1u;
    g0[1] = lds_offset(dst);
    g0[2] = (unsigned)ga;
    g0[3] = (unsigned)((ga >> 32) & 0x01ffffffull) | 0x80000000u;
    // D# group 1: data_size=2B, pad_enable, pad_interval=16DW, pad_amount=4DW,
    // tensor dims = remaining extent from tile start (OOB rows read as zero),
    // tile = 32(K) x 64(N) elements, dim0 stride = K elements.
    unsigned td0 = (unsigned)(K - kk);
    unsigned td1 = (unsigned)(N - n0);
    v8i g1;
    g1[0] = (int)((1u << 16) | (1u << 20) | (3u << 22) | (3u << 25));
    g1[1] = (int)((td0 & 0xffffu) << 16);
    g1[2] = (int)((td0 >> 16) | ((td1 & 0xffffu) << 16));
    g1[3] = (int)((td1 >> 16) | ((unsigned)BK << 16));
    g1[4] = (int)BN;
    g1[5] = (int)K;
    g1[6] = 0;
    g1[7] = 0;
    v4i gz = {};
#if __clang_major__ >= 23
    v8i z8 = {};
    __builtin_amdgcn_tensor_load_to_lds(g0, g1, gz, gz, z8, 0);
#else
    __builtin_amdgcn_tensor_load_to_lds(g0, g1, gz, gz, 0);
#endif
  };
#else
  const int br  = tid >> 2;
  const int bkc = (tid & 3) << 3;
#endif

  v8f acc[2][2] = {};

  // ---- prologue: stage tile 0 into buffer 0
  *(v8bf*)&sA[0][ar * LDT + akc] =
      *(const v8bf*)(A + (size_t)(m0 + ar) * K + akc);
  *(v8bf*)&sA[0][(ar + 64) * LDT + akc] =
      *(const v8bf*)(A + (size_t)(m0 + ar + 64) * K + akc);
#if HAVE_TDM
  if (wave == 0) {
    tdm_issue_B(0, &sB[0][0]);
    __builtin_amdgcn_s_wait_tensorcnt(0);
  }
#else
  {
    int gn = n0 + br;
    *(v8bf*)&sB[0][br * LDT + bkc] =
        (gn < N) ? *(const v8bf*)(W + (size_t)gn * K + bkc) : bzero8();
  }
#endif
  __syncthreads();

  const int nk = K / BK;
  int cur = 0;
  for (int ki = 0; ki < nk; ++ki) {
    const int kk = ki * BK;
    const bool more = (ki + 1 < nk);

    // ---- prefetch next tile (global loads / TDM overlap with WMMA below)
    v8bf pa0, pa1;
#if !HAVE_TDM
    v8bf pb0;
#endif
    if (more) {
      pa0 = *(const v8bf*)(A + (size_t)(m0 + ar) * K + kk + BK + akc);
      pa1 = *(const v8bf*)(A + (size_t)(m0 + ar + 64) * K + kk + BK + akc);
#if HAVE_TDM
      if (wave == 0) tdm_issue_B(kk + BK, &sB[cur ^ 1][0]);
#else
      int gn = n0 + br;
      pb0 = (gn < N) ? *(const v8bf*)(W + (size_t)gn * K + kk + BK + bkc)
                     : bzero8();
#endif
    }

    // ---- swizzled operand fetch per ISA 16-bit A/B layouts + 4 WMMAs
    V16 a[2], b[2];
#pragma unroll
    for (int mt = 0; mt < 2; ++mt) {
      const __bf16* p = &sA[cur][(wm * 32 + mt * 16 + lrow) * LDT + half * 8];
      a[mt].h[0] = *(const v8bf*)p;          // K = kb..kb+7
      a[mt].h[1] = *(const v8bf*)(p + 16);   // K = kb+16..kb+23
    }
#pragma unroll
    for (int nt = 0; nt < 2; ++nt) {
      const __bf16* p = &sB[cur][(wn * 32 + nt * 16 + lrow) * LDT + half * 8];
      b[nt].h[0] = *(const v8bf*)p;          // K = kb..kb+7
      b[nt].h[1] = *(const v8bf*)(p + 16);   // K = kb+16..kb+23
    }
#pragma unroll
    for (int mt = 0; mt < 2; ++mt)
#pragma unroll
      for (int nt = 0; nt < 2; ++nt)
        acc[mt][nt] = __builtin_amdgcn_wmma_f32_16x16x32_bf16(
            false, a[mt].v, false, b[nt].v, (short)0, acc[mt][nt], false, false);

    // ---- publish the prefetched tile into the other buffer
    if (more) {
      *(v8bf*)&sA[cur ^ 1][ar * LDT + akc] = pa0;
      *(v8bf*)&sA[cur ^ 1][(ar + 64) * LDT + akc] = pa1;
#if HAVE_TDM
      if (wave == 0) __builtin_amdgcn_s_wait_tensorcnt(0);
#else
      *(v8bf*)&sB[cur ^ 1][br * LDT + bkc] = pb0;
#endif
    }
    __syncthreads();
    cur ^= 1;
  }

  // ---- epilogue (C layout: lane holds col n=lane%16, rows r + 8*(lane/16))
  const int gm0 = m0 + wm * 32;
  const int gn0 = n0 + wn * 32;
#pragma unroll
  for (int mt = 0; mt < 2; ++mt) {
#pragma unroll
    for (int nt = 0; nt < 2; ++nt) {
      int n = gn0 + nt * 16 + lrow;
      if (n >= N) continue;
      float bia = bias[n];
#pragma unroll
      for (int r = 0; r < 8; ++r) {
        int m = gm0 + mt * 16 + half * 8 + r;
        float val = acc[mt][nt][r] + bia;
        if (EPI == 0) {                    // bf16 store (Q/K)
          outb[(size_t)m * N + n] = (__bf16)val;
        } else if (EPI == 1) {             // exact GELU -> bf16 (FFN1)
          float gl = 0.5f * val * (1.0f + erff(val * 0.70710678118654752f));
          outb[(size_t)m * N + n] = (__bf16)gl;
        } else if (EPI == 2) {             // residual add into f32 stream
          outf[(size_t)m * N + n] += val;
        } else if (EPI == 3) {             // f32 store (chord head)
          outf[(size_t)m * N + n] = val;
        } else {                           // EPI==4: V stored transposed (b,h,d,s)
          int bidx = m / S_DIM, s = m - (m / S_DIM) * S_DIM;
          int hh = n / D_DIM, d = n - (n / D_DIM) * D_DIM;
          outb[(((size_t)bidx * H_DIM + hh) * D_DIM + d) * S_DIM + s] = (__bf16)val;
        }
      }
    }
  }
}

// ---------------------------------------------------------------------------
// Flash-attention: block = 4 waves x 16 query rows; causal online softmax.
// q kept in registers (K=80 padded to 96); scores via 3 WMMAs per 16x16 tile;
// P transposed through a 1.25KB per-wave LDS buffer; P.V via 5 WMMAs per 32-K.
// No __syncthreads needed (per-wave LDS regions only).
// ---------------------------------------------------------------------------
__global__ __launch_bounds__(128)
void attn_kernel(const __bf16* __restrict__ q, const __bf16* __restrict__ k,
                 const __bf16* __restrict__ vT, __bf16* __restrict__ o) {
  const int b = blockIdx.z, h = blockIdx.y, qblk = blockIdx.x;
  const int tid = threadIdx.x, wave = tid >> 5, lane = tid & 31;
  const int half = lane >> 4, lrow = lane & 15;
  const int qrow0 = qblk * 64 + wave * 16;

  __shared__ __attribute__((aligned(16))) __bf16 pbuf[4][16][40];

  const __bf16* qbp = q + (size_t)b * S_DIM * E_DIM + h * D_DIM;
  const __bf16* kbp = k + (size_t)b * S_DIM * E_DIM + h * D_DIM;
  const __bf16* vbp = vT + ((size_t)b * H_DIM + h) * D_DIM * S_DIM;

  // q operand, A-layout: lane row = lane%16, K chunks per half
  V16 qa[3];
  {
    const __bf16* pr = qbp + (size_t)(qrow0 + lrow) * E_DIM;
#pragma unroll
    for (int c = 0; c < 3; ++c) {
      int d0 = c * 32 + half * 8, d1 = d0 + 16;
      qa[c].h[0] = *(const v8bf*)(pr + d0);                       // d0 <= 72 < 80
      qa[c].h[1] = (d1 < D_DIM) ? *(const v8bf*)(pr + d1) : bzero8();
    }
  }

  float M8[8], L8[8];
  v8f oacc[5] = {};
#pragma unroll
  for (int v = 0; v < 8; ++v) { M8[v] = -1e30f; L8[v] = 0.0f; }

  const float scale = 0.11180339887498949f;   // 1/sqrt(80)
  const int tmax = qrow0 >> 4;                // last key tile with any valid col
  const int npairs = (tmax >> 1) + 1;

  for (int tp = 0; tp < npairs; ++tp) {
    // ---- two 16x16 score tiles (K = d, padded to 96)
    v8f st[2] = {};
#pragma unroll
    for (int sub = 0; sub < 2; ++sub) {
      int t = tp * 2 + sub;
      if (t <= tmax) {
        const __bf16* kr = kbp + (size_t)(t * 16 + lrow) * E_DIM;
        v8f sa = {};
#pragma unroll
        for (int c = 0; c < 3; ++c) {
          V16 kv;
          int d0 = c * 32 + half * 8;
          kv.h[0] = (d0 < D_DIM) ? *(const v8bf*)(kr + d0) : bzero8();
          kv.h[1] = (d0 + 16 < D_DIM) ? *(const v8bf*)(kr + d0 + 16) : bzero8();
          sa = __builtin_amdgcn_wmma_f32_16x16x32_bf16(false, qa[c].v, false,
                                                       kv.v, (short)0, sa,
                                                       false, false);
        }
        st[sub] = sa;
      }
    }
    // ---- causal mask + online softmax over the 32-wide pair
    int col0 = tp * 32 + lrow, col1 = col0 + 16;
#pragma unroll
    for (int v = 0; v < 8; ++v) {
      int rowg = qrow0 + half * 8 + v;
      float x0 = (col0 <= rowg) ? st[0][v] * scale : -1e30f;
      float x1 = (col1 <= rowg) ? st[1][v] * scale : -1e30f;
      float mx = fmaxf(x0, x1);
#pragma unroll
      for (int msk = 1; msk < 16; msk <<= 1) mx = fmaxf(mx, __shfl_xor(mx, msk));
      float newM = fmaxf(M8[v], mx);
      float corr = __expf(M8[v] - newM);
      float p0 = __expf(x0 - newM);
      float p1 = __expf(x1 - newM);
      float ps = p0 + p1;
#pragma unroll
      for (int msk = 1; msk < 16; msk <<= 1) ps += __shfl_xor(ps, msk);
      L8[v] = L8[v] * corr + ps;
      M8[v] = newM;
#pragma unroll
      for (int dt = 0; dt < 5; ++dt) oacc[dt][v] *= corr;
      pbuf[wave][half * 8 + v][lrow]      = (__bf16)p0;   // C-layout -> A-layout
      pbuf[wave][half * 8 + v][16 + lrow] = (__bf16)p1;   // transpose via LDS
    }
    // ---- P(16x32) @ V(32x80): A from pbuf (swizzled), B from vT rows (d)
    V16 pa;
    {
      const __bf16* p = &pbuf[wave][lrow][half * 8];
      pa.h[0] = *(const v8bf*)p;
      pa.h[1] = *(const v8bf*)(p + 16);
    }
#pragma unroll
    for (int dt = 0; dt < 5; ++dt) {
      V16 vv;
      const __bf16* vr =
          vbp + (size_t)(dt * 16 + lrow) * S_DIM + tp * 32 + half * 8;
      vv.h[0] = *(const v8bf*)vr;
      vv.h[1] = *(const v8bf*)(vr + 16);
      oacc[dt] = __builtin_amdgcn_wmma_f32_16x16x32_bf16(
          false, pa.v, false, vv.v, (short)0, oacc[dt], false, false);
    }
  }

  // ---- normalize and store o as (b,s,e=h*80+d) bf16 for the Wo GEMM
#pragma unroll
  for (int dt = 0; dt < 5; ++dt)
#pragma unroll
    for (int v = 0; v < 8; ++v) {
      int s = qrow0 + half * 8 + v;
      int e = h * D_DIM + dt * 16 + lrow;
      o[((size_t)b * S_DIM + s) * E_DIM + e] = (__bf16)(oacc[dt][v] / L8[v]);
    }
}

// ---------------------------------------------------------------------------
// Host orchestration
// ---------------------------------------------------------------------------
static inline int cdiv(int a, int b) { return (a + b - 1) / b; }

extern "C" void kernel_launch(void* const* d_in, const int* in_sizes, int n_in,
                              void* d_out, int out_size, void* d_ws, size_t ws_size,
                              hipStream_t stream) {
  const int*   tok    = (const int*)  d_in[0];
  const float* tokemb = (const float*)d_in[1];
  const float* posemb = (const float*)d_in[2];
  const float* Wq  = (const float*)d_in[3];
  const float* bq  = (const float*)d_in[4];
  const float* Wk  = (const float*)d_in[5];
  const float* bk  = (const float*)d_in[6];
  const float* Wv  = (const float*)d_in[7];
  const float* bv  = (const float*)d_in[8];
  const float* Wo  = (const float*)d_in[9];
  const float* bo  = (const float*)d_in[10];
  const float* l1s = (const float*)d_in[11];
  const float* l1b = (const float*)d_in[12];
  const float* l2s = (const float*)d_in[13];
  const float* l2b = (const float*)d_in[14];
  const float* W1  = (const float*)d_in[15];
  const float* b1  = (const float*)d_in[16];
  const float* W2  = (const float*)d_in[17];
  const float* b2  = (const float*)d_in[18];
  const float* Wh  = (const float*)d_in[19];
  const float* bh  = (const float*)d_in[20];
  float* out = (float*)d_out;

  char* ws = (char*)d_ws;
  size_t off = 0;
  float*  xf  = (float*) (ws + off); off += (size_t)MTOK * E_DIM * 4;   // residual
  __bf16* hb  = (__bf16*)(ws + off); off += (size_t)MTOK * E_DIM * 2;   // LN out
  __bf16* qb  = (__bf16*)(ws + off); off += (size_t)MTOK * E_DIM * 2;
  __bf16* kb_ = (__bf16*)(ws + off); off += (size_t)MTOK * E_DIM * 2;
  __bf16* vTb = (__bf16*)(ws + off); off += (size_t)MTOK * E_DIM * 2;   // (b,h,d,s)
  __bf16* ob  = (__bf16*)(ws + off); off += (size_t)MTOK * E_DIM * 2;
  __bf16* ffb = qb;  // FFN hidden (16384x1920) aliases dead q/k/vT/o (exact fit)
  __bf16* xh  = (__bf16*)(ws + off); off += (size_t)MH * E_DIM * 2;
  __bf16* wqb = (__bf16*)(ws + off); off += (size_t)L_DIM * E_DIM * E_DIM * 2;
  __bf16* wkb = (__bf16*)(ws + off); off += (size_t)L_DIM * E_DIM * E_DIM * 2;
  __bf16* wvb = (__bf16*)(ws + off); off += (size_t)L_DIM * E_DIM * E_DIM * 2;
  __bf16* wob = (__bf16*)(ws + off); off += (size_t)L_DIM * E_DIM * E_DIM * 2;
  __bf16* w1b = (__bf16*)(ws + off); off += (size_t)L_DIM * F_DIM * E_DIM * 2;
  __bf16* w2b = (__bf16*)(ws + off); off += (size_t)L_DIM * E_DIM * F_DIM * 2;
  __bf16* whb = (__bf16*)(ws + off); off += (size_t)NCHORD * E_DIM * 2;

  auto cvt = [&](const float* s, __bf16* d, int n) {
    cvt_kernel<<<cdiv(n, 256), 256, 0, stream>>>(s, d, n);
  };
  cvt(Wq, wqb, L_DIM * E_DIM * E_DIM);
  cvt(Wk, wkb, L_DIM * E_DIM * E_DIM);
  cvt(Wv, wvb, L_DIM * E_DIM * E_DIM);
  cvt(Wo, wob, L_DIM * E_DIM * E_DIM);
  cvt(W1, w1b, L_DIM * F_DIM * E_DIM);
  cvt(W2, w2b, L_DIM * E_DIM * F_DIM);
  cvt(Wh, whb, NCHORD * E_DIM);

  embed_kernel<<<cdiv(MTOK * E_DIM, 256), 256, 0, stream>>>(tok, tokemb, posemb, xf);

  const dim3 gE(cdiv(E_DIM, BN), MTOK / BM);   // (8, 128)
  const dim3 gF(F_DIM / BN, MTOK / BM);        // (30, 128)
  for (int i = 0; i < L_DIM; ++i) {
    ln_kernel<<<MTOK / 8, 256, 0, stream>>>(xf, l1s + i * E_DIM, l1b + i * E_DIM, hb);
    gemm_bf16<0><<<gE, 256, 0, stream>>>(hb, wqb + (size_t)i * E_DIM * E_DIM,
                                         bq + i * E_DIM, qb, nullptr,
                                         MTOK, E_DIM, E_DIM);
    gemm_bf16<0><<<gE, 256, 0, stream>>>(hb, wkb + (size_t)i * E_DIM * E_DIM,
                                         bk + i * E_DIM, kb_, nullptr,
                                         MTOK, E_DIM, E_DIM);
    gemm_bf16<4><<<gE, 256, 0, stream>>>(hb, wvb + (size_t)i * E_DIM * E_DIM,
                                         bv + i * E_DIM, vTb, nullptr,
                                         MTOK, E_DIM, E_DIM);
    attn_kernel<<<dim3(S_DIM / 64, H_DIM, B_DIM), 128, 0, stream>>>(qb, kb_, vTb, ob);
    gemm_bf16<2><<<gE, 256, 0, stream>>>(ob, wob + (size_t)i * E_DIM * E_DIM,
                                         bo + i * E_DIM, nullptr, xf,
                                         MTOK, E_DIM, E_DIM);
    ln_kernel<<<MTOK / 8, 256, 0, stream>>>(xf, l2s + i * E_DIM, l2b + i * E_DIM, hb);
    gemm_bf16<1><<<gF, 256, 0, stream>>>(hb, w1b + (size_t)i * F_DIM * E_DIM,
                                         b1 + i * F_DIM, ffb, nullptr,
                                         MTOK, F_DIM, E_DIM);
    gemm_bf16<2><<<gE, 256, 0, stream>>>(ffb, w2b + (size_t)i * E_DIM * F_DIM,
                                         b2 + i * E_DIM, nullptr, xf,
                                         MTOK, E_DIM, F_DIM);
  }

  gather_odd_kernel<<<cdiv(MH * E_DIM, 256), 256, 0, stream>>>(xf, xh);
  const dim3 gH(cdiv(NCHORD, BN), MH / BM);    // (72, 64)
  gemm_bf16<3><<<gH, 256, 0, stream>>>(xh, whb, bh, nullptr, out,
                                       MH, NCHORD, E_DIM);
}
